// DynamicConvolution_77816217469047
// MI455X (gfx1250) — matrix operand order repaired and checked by reading, not compiled, not run on hardware
//
#include <hip/hip_runtime.h>

// ---------------------------------------------------------------------------
// DynamicConvolution for MI455X (gfx1250): bf16 WMMA implicit-GEMM conv with
// Tensor-Data-Mover (TDM) double-buffered LDS staging.
//   x:[16,256,56,56] f32  kernels:[8,256,256,3,3] f32  w1:[256,64] b1:[64]
//   w2:[64,8] b2:[8]   ->  out:[16,256,56,56] f32
// Workspace layout (bytes):
//   [0)        alphas       16*8 f32                (512 B)
//   [1024)     Aw  bf16     16*256*2304             (18,874,368 B)
//   [18875392) xpad bf16    16*58*58*256 (NHWC+pad) (27,557,888 B)
// ---------------------------------------------------------------------------

typedef __attribute__((ext_vector_type(16))) __bf16 v16bf;
typedef __attribute__((ext_vector_type(8)))  float  v8f;
typedef __attribute__((ext_vector_type(4)))  unsigned int u32x4;
typedef __attribute__((ext_vector_type(8)))  int  i32x8;
typedef __attribute__((ext_vector_type(4)))  int  i32x4;

#define B_BATCH 16
#define CHN 256
#define HW 3136          // 56*56
#define KTOT 2304        // 9*256
#define PADW 58
#define PADHW 3364       // 58*58

#define AW_OFF   1024
#define XPAD_OFF 18875392

static __device__ __forceinline__ __bf16 f2bf(float f) { return (__bf16)f; }

// ---------------------------------------------------------------------------
// 1) attention: global avg pool -> relu MLP -> softmax  (one block per sample)
// ---------------------------------------------------------------------------
__global__ __launch_bounds__(256)
void attn_kernel(const float* __restrict__ x, const float* __restrict__ w1,
                 const float* __restrict__ b1, const float* __restrict__ w2,
                 const float* __restrict__ b2, float* __restrict__ alphas)
{
    __shared__ float sp[CHN];
    __shared__ float sh[64];
    __shared__ float ss[8];
    const int b = blockIdx.x;
    const int tid = threadIdx.x;
    const int lane = tid & 31, wv = tid >> 5;

    for (int cc = 0; cc < 32; ++cc) {
        const int c = wv * 32 + cc;
        const float* px = x + ((size_t)b * CHN + c) * HW;
        float s = 0.f;
        for (int p = lane; p < HW; p += 32) s += px[p];
        #pragma unroll
        for (int off = 16; off > 0; off >>= 1) s += __shfl_xor(s, off);
        if (lane == 0) sp[c] = s * (1.0f / (float)HW);
    }
    __syncthreads();

    if (tid < 64) {
        float acc = b1[tid];
        for (int c = 0; c < CHN; ++c) acc += sp[c] * w1[c * 64 + tid];
        sh[tid] = acc > 0.f ? acc : 0.f;
    }
    __syncthreads();
    if (tid < 8) {
        float acc = b2[tid];
        for (int j = 0; j < 64; ++j) acc += sh[j] * w2[j * 8 + tid];
        ss[tid] = acc;
    }
    __syncthreads();
    if (tid == 0) {
        float mx = ss[0];
        for (int k = 1; k < 8; ++k) mx = fmaxf(mx, ss[k]);
        float e[8], sum = 0.f;
        for (int k = 0; k < 8; ++k) { e[k] = __expf(ss[k] - mx); sum += e[k]; }
        float inv = 1.0f / sum;
        for (int k = 0; k < 8; ++k) alphas[b * 8 + k] = e[k] * inv;
    }
}

// ---------------------------------------------------------------------------
// 2) mix kernels: Aw[b][o][tap*256+i] = sum_k alpha[b][k]*kernels[k][o][i][tap]
// ---------------------------------------------------------------------------
__global__ __launch_bounds__(256)
void mix_kernel(const float* __restrict__ kern, const float* __restrict__ alphas,
                __bf16* __restrict__ Aw)
{
    __shared__ float sa[8];
    const int b = blockIdx.z;
    const int tid = threadIdx.x;
    if (tid < 8) sa[tid] = alphas[b * 8 + tid];
    __syncthreads();

    const int idx = blockIdx.x * 256 + tid;
    const int o = idx / KTOT;
    const int k = idx - o * KTOT;
    const int tap = k >> 8;
    const int i = k & 255;
    const size_t base = ((size_t)o * CHN + i) * 9 + tap;
    float acc = 0.f;
    #pragma unroll
    for (int kk = 0; kk < 8; ++kk)
        acc += sa[kk] * kern[(size_t)kk * CHN * CHN * 9 + base];
    Aw[((size_t)b * CHN + o) * KTOT + k] = f2bf(acc);
}

// ---------------------------------------------------------------------------
// 3) pad+transpose: x NCHW f32 -> xpad [b][58][58][256] bf16 (border pre-zeroed)
// ---------------------------------------------------------------------------
__global__ __launch_bounds__(256)
void pad_kernel(const float* __restrict__ x, __bf16* __restrict__ xpad)
{
    __shared__ float tile[32][133];
    const int b  = blockIdx.z;
    const int c0 = blockIdx.y * 32;
    const int p0 = blockIdx.x * 128;
    const int tid = threadIdx.x;

    const float* xs = x + ((size_t)b * CHN + c0) * HW;
    #pragma unroll
    for (int r = 0; r < 16; ++r) {
        const int idx = r * 256 + tid;
        const int c = idx >> 7, p = idx & 127;
        tile[c][p] = (p0 + p < HW) ? xs[(size_t)c * HW + p0 + p] : 0.f;
    }
    __syncthreads();
    #pragma unroll
    for (int r = 0; r < 16; ++r) {
        const int idx = r * 256 + tid;
        const int p = idx >> 5, c = idx & 31;
        const int pos = p0 + p;
        if (pos < HW) {
            const int py = pos / 56, px2 = pos - py * 56;
            const size_t dst = ((size_t)b * PADHW + (py + 1) * PADW + (px2 + 1)) * CHN
                             + c0 + c;
            xpad[dst] = f2bf(tile[c][p]);
        }
    }
}

// ---------------------------------------------------------------------------
// 4) conv implicit GEMM: TDM double-buffered staging + v_wmma_f32_16x16x32_bf16
// C tile [128 o][128 pos], pos-tile = 8(y) x 16(x) rectangle.
// LDS buffer s (s=0,1) at byte s*20480: A tile [128 rows][64B]+16B pad (10240 B)
//                                       B tile [128 pos ][64B]+16B pad (10240 B)
// K loop: 72 chunks of 32 (k = tap*256 + channel).
// ---------------------------------------------------------------------------
#define AST 80          // LDS row stride produced by TDM pad (64B + 16B)
#define LBUF 20480      // one double-buffer slice (A 10240 + B 10240)

union Frag { u32x4 u[2]; v16bf v; };

__global__ __launch_bounds__(256)
void conv_kernel(const __bf16* __restrict__ Aw, const __bf16* __restrict__ xpad,
                 float* __restrict__ out)
{
    __shared__ __align__(16) unsigned char lds[2 * LBUF];

    // Escape the LDS base address into an asm that may write memory: the TDM
    // engine is the only writer of `lds`, invisible to LLVM.  Without the
    // escape, "no visible stores" lets every ds_load fold to undef; with it,
    // the per-iteration memory clobber below forces real reloads while the
    // pointer stays a traceable addrspace(3) value (ds_load_b128, not flat).
    asm volatile("" : : "v"((void*)lds) : "memory");

    const int b     = blockIdx.z;
    const int mBase = blockIdx.y * 128;
    const int ty    = blockIdx.x >> 2;          // 0..6  -> y0 = 8*ty
    const int tx    = blockIdx.x & 3;           // 0..3  -> x0 = 16*tx (48 partial)
    const int y0    = ty * 8, x0 = tx * 16;
    const int tid   = threadIdx.x;
    const int lane  = tid & 31, wv = tid >> 5;
    const int wm    = wv >> 2,  wn = wv & 3;    // 2 x 4 wave grid
    const int half  = lane >> 4, lr = lane & 15;

    // 64-bit global byte addresses for the TDM descriptors
    const unsigned long long aBase =
        (unsigned long long)(const void*)(Aw + ((size_t)b * CHN + mBase) * KTOT);
    const unsigned long long xBase =
        (unsigned long long)(const void*)(xpad + (size_t)b * PADHW * CHN);

    const i32x4 z4 = { 0, 0, 0, 0 };
    const i32x8 z8 = { 0, 0, 0, 0, 0, 0, 0, 0 };

    // issue TDM loads of K-chunk kc into LDS buffer s (wave-uniform values)
    auto issue = [&](int kc, int s) {
        // ---- A tile: 2D, 32 bf16 x 128 rows, row stride 2304 elems ----
        const unsigned long long ga = aBase + (unsigned long long)kc * 64;
        u32x4 g0a = { 1u, (unsigned)(s * LBUF),
                      (unsigned)ga, (unsigned)(ga >> 32) | 0x80000000u };
        i32x8 g1a = { 0x06D10000,              // data_size=2B, pad 16B per 64B
                      0x09000000,              // tensor_dim0 = 2304 (<<16)
                      0x10000000,              // tensor_dim1 = 4096 (<<16)
                      0x00200000,              // tile_dim0 = 32 (<<16)
                      128,                     // tile_dim1 = 128, tile_dim2 = 0
                      2304,                    // tensor_dim0_stride
                      0, 0 };
        __builtin_amdgcn_tensor_load_to_lds(g0a, g1a, z4, z4, z8, 0);

        // ---- B tile: 3D, 32 ch x 16 x-pixels x 8 y-rows into padded NHWC ----
        const int tap = kc >> 3;
        const int dy = tap / 3, dx = tap - dy * 3;
        const int c0 = (kc & 7) * 32;
        const unsigned long long gb = xBase +
            2ull * ((unsigned long long)((y0 + dy) * PADW + (x0 + dx)) * CHN + c0);
        const int xrem = PADW - (x0 + dx);     // OOB x -> TDM zero-fill
        const int yrem = PADW - (y0 + dy);
        u32x4 g0b = { 1u, (unsigned)(s * LBUF + 10240),
                      (unsigned)gb, (unsigned)(gb >> 32) | 0x80000000u };
        i32x8 g1b = { 0x06D10000,
                      0x01000000,              // tensor_dim0 = 256 (<<16)
                      xrem << 16,              // tensor_dim1 = remaining x extent
                      0x00200000,              // tile_dim0 = 32 (<<16)
                      16 | (8 << 16),          // tile_dim1 = 16, tile_dim2 = 8
                      256,                     // dim0 stride (between x-pixels)
                      14848 << 16,             // dim1 stride = 58*256 (low16<<16)
                      0 };                     // dim1 stride hi = 0
        i32x4 g2b = { yrem, 0, 0, 0 };         // tensor_dim2 = remaining y extent
        __builtin_amdgcn_tensor_load_to_lds(g0b, g1b, g2b, z4, z8, 0);
    };

    v8f acc[4][2] = {};

    if (wv == 0) issue(0, 0);
    for (int kc = 0; kc < 72; ++kc) {
        const int s = kc & 1;
        __syncthreads();                       // buf s^1 reads done -> reusable
        if (wv == 0) {
            if (kc + 1 < 72) { issue(kc + 1, s ^ 1);
                               __builtin_amdgcn_s_wait_tensorcnt((short)2); }
            else             { __builtin_amdgcn_s_wait_tensorcnt((short)0); }
        }
        __syncthreads();                       // buf s ready for all waves

        // TDM has written LDS behind the compiler's back: force reloads.
        asm volatile("" ::: "memory");

        const unsigned char* As = lds + s * LBUF;
        const unsigned char* Bs = As + 10240;

        Frag fa[4], fb[2];
        #pragma unroll
        for (int ss2 = 0; ss2 < 4; ++ss2) {    // A: lane half picks K sub-ranges
            const unsigned char* p = As + (wm * 64 + ss2 * 16 + lr) * AST + half * 16;
            fa[ss2].u[0] = *(const u32x4*)p;
            fa[ss2].u[1] = *(const u32x4*)(p + 32);
        }
        #pragma unroll
        for (int t = 0; t < 2; ++t) {          // B: lane = column, K table order
            const unsigned char* p = Bs + (wn * 32 + t * 16 + lr) * AST + half * 16;
            fb[t].u[0] = *(const u32x4*)p;
            fb[t].u[1] = *(const u32x4*)(p + 32);
        }
        #pragma unroll
        for (int ss2 = 0; ss2 < 4; ++ss2)
            #pragma unroll
            for (int t = 0; t < 2; ++t)
                acc[ss2][t] = __builtin_amdgcn_wmma_f32_16x16x32_bf16(
                    false, fa[ss2].v, false, fb[t].v,
                    (short)0, acc[ss2][t], false, false);
    }

    // epilogue: subtile ns -> image row y0+ns, columns x0..x0+15
    #pragma unroll
    for (int ss2 = 0; ss2 < 4; ++ss2) {
        #pragma unroll
        for (int t = 0; t < 2; ++t) {
            const int xcol = x0 + lr;
            const int yrow = y0 + wn * 2 + t;
            if (xcol < 56) {
                #pragma unroll
                for (int r = 0; r < 8; ++r) {
                    const int row = mBase + wm * 64 + ss2 * 16 + r + 8 * half;
                    out[((size_t)b * CHN + row) * HW + yrow * 56 + xcol]
                        = acc[ss2][t][r];
                }
            }
        }
    }
}

// ---------------------------------------------------------------------------
extern "C" void kernel_launch(void* const* d_in, const int* in_sizes, int n_in,
                              void* d_out, int out_size, void* d_ws, size_t ws_size,
                              hipStream_t stream)
{
    const float* x    = (const float*)d_in[0];
    const float* kern = (const float*)d_in[1];
    const float* w1   = (const float*)d_in[2];
    const float* b1   = (const float*)d_in[3];
    const float* w2   = (const float*)d_in[4];
    const float* b2   = (const float*)d_in[5];
    float* out = (float*)d_out;

    char* ws = (char*)d_ws;
    float*  alphas = (float*)ws;
    __bf16* Aw     = (__bf16*)(ws + AW_OFF);
    __bf16* xpad   = (__bf16*)(ws + XPAD_OFF);

    (void)hipMemsetAsync(xpad, 0,
                         (size_t)B_BATCH * PADHW * CHN * sizeof(__bf16), stream);

    attn_kernel<<<dim3(B_BATCH), 256, 0, stream>>>(x, w1, b1, w2, b2, alphas);
    mix_kernel <<<dim3(2304, 1, B_BATCH), 256, 0, stream>>>(kern, alphas, Aw);
    pad_kernel <<<dim3(25, 8, B_BATCH), 256, 0, stream>>>(x, xpad);
    conv_kernel<<<dim3(28, 2, B_BATCH), 256, 0, stream>>>(Aw, xpad, out);
}
